// W4A8Layer_84086869721792
// MI455X (gfx1250) — 8-bit matrix operand order repaired and checked by reading, compiled, not run on hardware
//
#include <hip/hip_runtime.h>
#include <stdint.h>

typedef int   v8i __attribute__((ext_vector_type(8)));
typedef float v8f __attribute__((ext_vector_type(8)));

#define M_DIM 512
#define K_DIM 4096
#define N_DIM 12288
#define NGROUPS 32

#define BM 128
#define BN 128
#define KC 128

#if __has_builtin(__builtin_amdgcn_global_load_async_to_lds_b128)
#define HAS_ASYNC 1
#else
#define HAS_ASYNC 0
#endif

typedef int v4i_t __attribute__((vector_size(4 * sizeof(int))));
typedef __attribute__((address_space(1))) v4i_t* g_v4i_p;
typedef __attribute__((address_space(3))) v4i_t* l_v4i_p;

__device__ __forceinline__ void stage_b128(const int8_t* g, int8_t* l) {
#if HAS_ASYNC
    __builtin_amdgcn_global_load_async_to_lds_b128(
        (g_v4i_p)(uintptr_t)g,
        (l_v4i_p)(uintptr_t)(unsigned)(uintptr_t)l,
        0, 0);
#else
    *(int4*)l = *(const int4*)g;
#endif
}

__device__ __forceinline__ void wait_async() {
#if HAS_ASYNC
#if __has_builtin(__builtin_amdgcn_s_wait_asynccnt)
    __builtin_amdgcn_s_wait_asynccnt(0);
#else
    asm volatile("s_wait_asynccnt 0x0" ::: "memory");
#endif
#endif
}

__global__ __launch_bounds__(256)
void w4a8_wmma_kernel(const int8_t* __restrict__ A,
                      const int8_t* __restrict__ W,
                      const float*  __restrict__ gscale,
                      const float*  __restrict__ cscale,
                      const float*  __restrict__ tscale,
                      float*        __restrict__ out)
{
    __shared__ int8_t As[2][BM * KC];
    __shared__ int8_t Bs[2][BN * KC];

    const int tid  = threadIdx.x;
    const int lane = tid & 31;
    const int wave = tid >> 5;
    const int wm   = wave & 3;
    const int wn   = wave >> 2;

    const int blockN = blockIdx.x * BN;
    const int blockM = blockIdx.y * BM;

    const int ln  = lane & 15;
    const int lhi = lane >> 4;

    const int arow = tid >> 3;
    const int acol = (tid & 7) * 16;
    const int kr4  = (tid >> 3) * 4;
    const int bcol = (tid & 7) * 16;

    v8f facc[2][4] = {};

    int4 bregs[4];

    auto stageA = [&](int g, int buf) {
        const int k0 = g * KC;
        #pragma unroll
        for (int r = 0; r < 4; ++r) {
            const int row = arow + r * 32;
            stage_b128(A + (size_t)(blockM + row) * K_DIM + k0 + acol,
                       &As[buf][row * KC + acol]);
        }
    };
    auto loadBregs = [&](int g) {
        const int k0 = g * KC;
        #pragma unroll
        for (int r = 0; r < 4; ++r)
            bregs[r] = *(const int4*)(W + (size_t)(k0 + kr4 + r) * N_DIM + blockN + bcol);
    };
    auto transB = [&](int buf) {
        const unsigned* r0 = (const unsigned*)&bregs[0];
        const unsigned* r1 = (const unsigned*)&bregs[1];
        const unsigned* r2 = (const unsigned*)&bregs[2];
        const unsigned* r3 = (const unsigned*)&bregs[3];
        #pragma unroll
        for (int n = 0; n < 16; ++n) {
            const int w = n >> 2, j = n & 3;
#if __has_builtin(__builtin_amdgcn_perm)
            const unsigned sel = 0x04000400u + (unsigned)j * 0x01010101u;
            const unsigned lo = __builtin_amdgcn_perm(r1[w], r0[w], sel);
            const unsigned hi = __builtin_amdgcn_perm(r3[w], r2[w], sel);
            const unsigned d  = __builtin_amdgcn_perm(hi, lo, 0x05040100u);
#else
            const unsigned d = ((r0[w] >> (8 * j)) & 0xffu)
                             | (((r1[w] >> (8 * j)) & 0xffu) << 8)
                             | (((r2[w] >> (8 * j)) & 0xffu) << 16)
                             | (((r3[w] >> (8 * j)) & 0xffu) << 24);
#endif
            *(unsigned*)&Bs[buf][(bcol + n) * KC + kr4] = d;
        }
    };

    stageA(0, 0);
    loadBregs(0);
    transB(0);
    wait_async();
    __syncthreads();

    #pragma unroll 1
    for (int g = 0; g < NGROUPS; ++g) {
        const int cur = g & 1, nxt = cur ^ 1;
        const bool more = (g + 1 < NGROUPS);

        if (more) {
            stageA(g + 1, nxt);
            loadBregs(g + 1);
        }
        if (g + 2 < NGROUPS) {
            __builtin_prefetch(W + (size_t)((g + 2) * KC + kr4) * N_DIM + blockN + bcol, 0, 1);
        }

        v8i iacc[2][4] = {};
        #pragma unroll
        for (int ks = 0; ks < 2; ++ks) {
            v8i af[2];
            #pragma unroll
            for (int mt = 0; mt < 2; ++mt) {
                const int8_t* pa = &As[cur][(wm * 32 + mt * 16 + ln) * KC + ks * 64 + lhi * 16];
                int2 q0 = *(const int2*)(pa +  0);
                int2 q1 = *(const int2*)(pa +  8);
                int2 q2 = *(const int2*)(pa + 32);
                int2 q3 = *(const int2*)(pa + 40);
                af[mt][0] = q0.x; af[mt][1] = q0.y; af[mt][2] = q1.x; af[mt][3] = q1.y;
                af[mt][4] = q2.x; af[mt][5] = q2.y; af[mt][6] = q3.x; af[mt][7] = q3.y;
            }
            #pragma unroll
            for (int t = 0; t < 4; ++t) {
                const int8_t* pb = &Bs[cur][(wn * 64 + t * 16 + ln) * KC + ks * 64 + lhi * 16];
                v8i bf;
                int4 b0 = *(const int4*)(pb);
                int4 b1 = *(const int4*)(pb + 32);
                bf[0] = b0.x; bf[1] = b0.y; bf[2] = b0.z; bf[3] = b0.w;
                bf[4] = b1.x; bf[5] = b1.y; bf[6] = b1.z; bf[7] = b1.w;
                #pragma unroll
                for (int mt = 0; mt < 2; ++mt) {
                    iacc[mt][t] = __builtin_amdgcn_wmma_i32_16x16x64_iu8(
                        true, af[mt], true, bf, iacc[mt][t], false, false);
                }
            }
        }

        #pragma unroll
        for (int t = 0; t < 4; ++t) {
            const int colN = blockN + wn * 64 + t * 16 + ln;
            const float gsv = gscale[(size_t)g * N_DIM + colN];
            #pragma unroll
            for (int mt = 0; mt < 2; ++mt)
                #pragma unroll
                for (int e = 0; e < 8; ++e)
                    facc[mt][t][e] += (float)iacc[mt][t][e] * gsv;
        }

        if (more) transB(nxt);
        wait_async();
        __syncthreads();
    }

    float ts[2][8];
    #pragma unroll
    for (int mt = 0; mt < 2; ++mt)
        #pragma unroll
        for (int e = 0; e < 8; ++e)
            ts[mt][e] = tscale[blockM + wm * 32 + mt * 16 + lhi * 8 + e];

    #pragma unroll
    for (int t = 0; t < 4; ++t) {
        const int colN = blockN + wn * 64 + t * 16 + ln;
        const float cs = cscale[colN];
        #pragma unroll
        for (int mt = 0; mt < 2; ++mt) {
            #pragma unroll
            for (int e = 0; e < 8; ++e) {
                const int row = blockM + wm * 32 + mt * 16 + lhi * 8 + e;
                out[(size_t)row * N_DIM + colN] = facc[mt][t][e] * ts[mt][e] * cs;
            }
        }
    }
}

extern "C" void kernel_launch(void* const* d_in, const int* in_sizes, int n_in,
                              void* d_out, int out_size, void* d_ws, size_t ws_size,
                              hipStream_t stream) {
    const int8_t* a   = (const int8_t*)d_in[0];
    const int8_t* w   = (const int8_t*)d_in[1];
    const float*  gs  = (const float*)d_in[2];
    const float*  cs  = (const float*)d_in[3];
    const float*  tsc = (const float*)d_in[4];
    float* out = (float*)d_out;

    dim3 grid(N_DIM / BN, M_DIM / BM);
    dim3 block(256);
    hipLaunchKernelGGL(w4a8_wmma_kernel, grid, block, 0, stream,
                       a, w, gs, cs, tsc, out);
}
